// DeformableInceptionModule_44375602102287
// MI455X (gfx1250) — matrix operand order repaired and checked by reading, not compiled, not run on hardware
//
#include <hip/hip_runtime.h>

// Problem constants (match reference)
#define HH   128
#define WWID 128
#define HW   (HH * WWID)
#define NB   2
#define CIN  64
#define COUT 32
#define NPIX (NB * HW)
// zero-padded (halo = 3 = max pad) channels-last input for im2col
#define HALO 3
#define HP   (HH + 2 * HALO)
#define WP   (WWID + 2 * HALO)

typedef __attribute__((ext_vector_type(16))) __bf16 v16bf;
typedef __attribute__((ext_vector_type(8)))  float  v8f;
typedef __attribute__((ext_vector_type(4)))  float  f4;
typedef __attribute__((ext_vector_type(4)))  int    i4;

__device__ __forceinline__ __bf16 f2bf(float f) { return (__bf16)f; }

// 16-bit A-matrix 16x32 packing (contiguous K, matching B):
// lanes 0-15: elements 0..15 hold K=0..15
// lanes16-31: elements 0..15 hold K=16..31
__device__ __forceinline__ int amapK(int e, int half) {
  return half * 16 + e;
}
// 16-bit B-matrix 32x16 as loaded here (contiguous K, not AMD's table order): lanes 0-15 hold K=0..15, lanes 16-31 K=16..31
// -> with K = t*64 + c, a lane's 16 B elements are CONTIGUOUS channels.

// ---------------------------------------------------------------------------
// One-shot repacks (tiny vs 23.3 TB/s HBM)
// ---------------------------------------------------------------------------

// x [B,C,H,W] f32 -> xTp [B,HP,WP,C] bf16 (zero halo) and xTf [B,H,W,C] f32
__global__ void __launch_bounds__(256)
pad_transpose_kernel(const float* __restrict__ x, float* __restrict__ xTf,
                     __bf16* __restrict__ xTp) {
  int tid = blockIdx.x * 256 + threadIdx.x;
  if (tid >= NB * HP * WP * CIN) return;
  int c  = tid & 63;
  int r  = tid >> 6;
  int xp = r % WP;
  int r2 = r / WP;
  int yp = r2 % HP;
  int b  = r2 / HP;
  int y  = yp - HALO, xq = xp - HALO;
  bool in = (y >= 0) & (y < HH) & (xq >= 0) & (xq < WWID);
  float v = 0.f;
  if (in) v = x[(b * CIN + c) * HW + y * WWID + xq];
  xTp[tid] = f2bf(v);
  if (in) xTf[((size_t)(b * HW) + y * WWID + xq) * CIN + c] = v;
}

// Pack offset+mask weights into WMMA A-register layout, K = t*64 + c.
// Ap element index = ((mtile*(KTOT/32)+chunk)*32 + lane)*16 + e
template <int KS>
__global__ void __launch_bounds__(256)
wpack_kernel(const float* __restrict__ w_off, const float* __restrict__ w_msk,
             __bf16* __restrict__ Ap) {
  constexpr int T = KS * KS, MCH = 3 * T, KTOT = CIN * T, MT = (MCH + 15) / 16;
  int tid = blockIdx.x * 256 + threadIdx.x;
  if (tid >= MT * KTOT * 16) return;
  int e = tid & 15;
  int lane = (tid >> 4) & 31;
  int cidx = tid >> 9;
  int chunk = cidx % (KTOT / 32);
  int mtile = cidx / (KTOT / 32);
  int half = lane >> 4;
  int m = mtile * 16 + (lane & 15);
  int K = chunk * 32 + amapK(e, half);
  int t = K >> 6, c = K & 63;          // K = t*64 + c
  float v = 0.f;
  if (m < MCH)
    v = (m < 2 * T) ? w_off[m * KTOT + c * T + t]
                    : w_msk[(m - 2 * T) * KTOT + c * T + t];
  Ap[tid] = f2bf(v);
}

// Pack filt [COUT, C, T] into WMMA A layout, K = t*64 + c. (2 M-tiles exact)
template <int KS>
__global__ void __launch_bounds__(256)
fpack_kernel(const float* __restrict__ filt, __bf16* __restrict__ Fp) {
  constexpr int T = KS * KS, KTOT = CIN * T;
  int tid = blockIdx.x * 256 + threadIdx.x;
  if (tid >= 2 * KTOT * 16) return;
  int e = tid & 15;
  int lane = (tid >> 4) & 31;
  int cidx = tid >> 9;
  int chunk = cidx % (KTOT / 32);
  int mtile = cidx / (KTOT / 32);
  int half = lane >> 4;
  int m = mtile * 16 + (lane & 15);
  int K = chunk * 32 + amapK(e, half);
  int t = K >> 6, c = K & 63;
  Fp[tid] = f2bf(filt[m * KTOT + c * T + t]);
}

// ---------------------------------------------------------------------------
// Kernel 1: fused offset+mask conv as WMMA GEMM over packed operands.
// Wave = one 16-ch M-tile x 32 pixels (2 N-tiles sharing the A tile).
// Padded input -> branch-free fully unrolled tap loop:
//   per 32-K chunk: 1x A b128x2 + 2x B b128x2 + 2x wmma.
// ---------------------------------------------------------------------------
template <int KS>
__global__ void __launch_bounds__(256)
offmask_conv_kernel(const __bf16* __restrict__ xTp,
                    const v16bf* __restrict__ Ap,
                    const float* __restrict__ b_off,
                    const float* __restrict__ b_msk,
                    float* __restrict__ om) {
  constexpr int T = KS * KS, PAD = KS / 2, MCH = 3 * T, KTOT = CIN * T;
  constexpr int MT = (MCH + 15) / 16, NCH = KTOT / 32;

  const int lane = threadIdx.x & 31;
  const int wave = threadIdx.x >> 5;
  const int tile = blockIdx.x * 8 + wave;   // grids are exact multiples of 8
  const int mtile = tile % MT;
  const int ntile = tile / MT;
  const int half = lane >> 4;
  const int l15 = lane & 15;

  const int p0  = ntile * 32 + l15;        // pixels p0 and p0+16, same row
  const int pb  = p0 >> 14;                // / HW
  const int sp0 = p0 & (HW - 1);
  const int py  = sp0 >> 7;
  const int px0 = sp0 & 127;

  const v16bf* Aw = Ap + (size_t)mtile * NCH * 32 + lane;
  const __bf16* xb0 =
      xTp + ((size_t)(pb * HP + py + (HALO - PAD)) * WP + px0 + (HALO - PAD)) * CIN;
  const __bf16* xb1 = xb0 + 16 * CIN;

  v8f acc0 = {}, acc1 = {};
#pragma unroll
  for (int t = 0; t < T; ++t) {
    const int ti = t / KS, tj = t % KS;    // constants after unroll
    const v16bf* bp0 = (const v16bf*)(xb0 + (ti * WP + tj) * CIN);
    const v16bf* bp1 = (const v16bf*)(xb1 + (ti * WP + tj) * CIN);
#pragma unroll
    for (int cb = 0; cb < 2; ++cb) {
      v16bf a  = Aw[(t * 2 + cb) * 32];
      v16bf b0 = bp0[cb * 2 + half];
      v16bf b1 = bp1[cb * 2 + half];
      acc0 = __builtin_amdgcn_wmma_f32_16x16x32_bf16(false, a, false, b0,
                                                     (short)0, acc0, false, false);
      acc1 = __builtin_amdgcn_wmma_f32_16x16x32_bf16(false, a, false, b1,
                                                     (short)0, acc1, false, false);
    }
  }

  // C/D layout: row = v + 8*half, col = l15
  const int spbase = py * WWID + px0;
#pragma unroll
  for (int v = 0; v < 8; ++v) {
    int ch = mtile * 16 + v + 8 * half;
    if (ch < MCH) {
      float bias = (ch < 2 * T) ? b_off[ch] : b_msk[ch - 2 * T];
      float* o = om + ((size_t)pb * MCH + ch) * HW + spbase;
      o[0]  = acc0[v] + bias;
      o[16] = acc1[v] + bias;
    }
  }
}

// ---------------------------------------------------------------------------
// Kernel 2: deformable conv. 64 threads (2 waves) per 32-pixel tile.
// Phase 1: bilinear corner indices + mask-folded weights -> LDS.
// Phase 2: each wave handles 16 pixels x BOTH M-tiles (COUT=32), so the
// expensive gathered B tile is built once and fed to 2 wmma.
// ---------------------------------------------------------------------------
template <int KS>
__global__ void __launch_bounds__(64)
deform_gemm_kernel(const float* __restrict__ xTf,
                   const float* __restrict__ om,
                   const v16bf* __restrict__ Fp,
                   float* __restrict__ out, int part) {
  constexpr int T = KS * KS, PAD = KS / 2, KTOT = CIN * T;
  constexpr int NCH = KTOT / 32;
  constexpr int NE = T * 32;
  __shared__ int   s_idx[NE * 4];
  __shared__ float s_w[NE * 4];

  const int tilebase = blockIdx.x * 32;

  // ---- Phase 1 ----
  for (int i = threadIdx.x; i < NE; i += 64) {
    int t = i >> 5;
    int pl = i & 31;
    int p = tilebase + pl;
    int pb = p >> 14;
    int sp = p & (HW - 1);
    int py = sp >> 7;
    int px = sp & 127;
    const float dy = om[((pb * 3 * T) + 2 * t) * HW + sp];
    const float dx = om[((pb * 3 * T) + 2 * t + 1) * HW + sp];
    const float mk = om[((pb * 3 * T) + 2 * T + t) * HW + sp];
    float ys = (float)(py - PAD + t / KS) + dy;
    float xs = (float)(px - PAD + t % KS) + dx;
    float y0 = floorf(ys), x0 = floorf(xs);
    float wy1 = ys - y0, wx1 = xs - x0;
    float cw[4] = {(1.f - wy1) * (1.f - wx1), (1.f - wy1) * wx1,
                   wy1 * (1.f - wx1),         wy1 * wx1};
    float cy[4] = {y0, y0, y0 + 1.f, y0 + 1.f};
    float cx[4] = {x0, x0 + 1.f, x0, x0 + 1.f};
#pragma unroll
    for (int q = 0; q < 4; ++q) {
      bool valid = (cy[q] >= 0.f) && (cy[q] <= (float)(HH - 1)) &&
                   (cx[q] >= 0.f) && (cx[q] <= (float)(WWID - 1));
      int yi = (int)fminf(fmaxf(cy[q], 0.f), (float)(HH - 1));
      int xi = (int)fminf(fmaxf(cx[q], 0.f), (float)(WWID - 1));
      s_idx[i * 4 + q] = pb * HW + yi * WWID + xi;   // spatial idx (chan-last)
      s_w[i * 4 + q]   = valid ? cw[q] * mk : 0.f;   // fold mask modulation
    }
  }
  __syncthreads();

  // ---- Phase 2 ----
  const int wave = threadIdx.x >> 5;
  const int lane = threadIdx.x & 31;
  const int half = lane >> 4;
  const int l15 = lane & 15;

  const v16bf* Aw0 = Fp + lane;                       // M-tile 0
  const v16bf* Aw1 = Fp + (size_t)NCH * 32 + lane;    // M-tile 1
  const int entb = wave * 16 + l15;

  v8f acc0 = {}, acc1 = {};
  for (int t = 0; t < T; ++t) {
    const int ent = (t * 32 + entb) * 4;
    const i4 ix = *(const i4*)&s_idx[ent];   // ds_load_b128
    const f4 wq = *(const f4*)&s_w[ent];     // ds_load_b128
    const float* q0 = xTf + (size_t)ix.x * CIN;
    const float* q1 = xTf + (size_t)ix.y * CIN;
    const float* q2 = xTf + (size_t)ix.z * CIN;
    const float* q3 = xTf + (size_t)ix.w * CIN;
#pragma unroll
    for (int cb = 0; cb < 2; ++cb) {
      const int coff = cb * 32 + half * 16;
      v16bf b;
#pragma unroll
      for (int r = 0; r < 4; ++r) {
        f4 c0 = *(const f4*)(q0 + coff + r * 4);
        f4 c1 = *(const f4*)(q1 + coff + r * 4);
        f4 c2 = *(const f4*)(q2 + coff + r * 4);
        f4 c3 = *(const f4*)(q3 + coff + r * 4);
        f4 s = wq.x * c0 + wq.y * c1 + wq.z * c2 + wq.w * c3;
        b[4 * r + 0] = f2bf(s.x);
        b[4 * r + 1] = f2bf(s.y);
        b[4 * r + 2] = f2bf(s.z);
        b[4 * r + 3] = f2bf(s.w);
      }
      v16bf a0 = Aw0[(t * 2 + cb) * 32];
      v16bf a1 = Aw1[(t * 2 + cb) * 32];
      acc0 = __builtin_amdgcn_wmma_f32_16x16x32_bf16(false, a0, false, b,
                                                     (short)0, acc0, false, false);
      acc1 = __builtin_amdgcn_wmma_f32_16x16x32_bf16(false, a1, false, b,
                                                     (short)0, acc1, false, false);
    }
  }

  const int p = tilebase + entb;
  const int pb = p >> 14;
  const int sp = p & (HW - 1);
#pragma unroll
  for (int v = 0; v < 8; ++v) {
    int o = v + 8 * half;
    out[((size_t)pb * 96 + part + o) * HW + sp]      = acc0[v];
    out[((size_t)pb * 96 + part + 16 + o) * HW + sp] = acc1[v];
  }
}

// ---------------------------------------------------------------------------
extern "C" void kernel_launch(void* const* d_in, const int* in_sizes, int n_in,
                              void* d_out, int out_size, void* d_ws, size_t ws_size,
                              hipStream_t stream) {
  (void)in_sizes; (void)n_in; (void)out_size; (void)ws_size;
  const float* x      = (const float*)d_in[0];
  const float* w_off1 = (const float*)d_in[1];
  const float* b_off1 = (const float*)d_in[2];
  const float* w_msk1 = (const float*)d_in[3];
  const float* b_msk1 = (const float*)d_in[4];
  const float* filt1  = (const float*)d_in[5];
  const float* w_off2 = (const float*)d_in[6];
  const float* b_off2 = (const float*)d_in[7];
  const float* w_msk2 = (const float*)d_in[8];
  const float* b_msk2 = (const float*)d_in[9];
  const float* filt2  = (const float*)d_in[10];
  const float* w_off3 = (const float*)d_in[11];
  const float* b_off3 = (const float*)d_in[12];
  const float* w_msk3 = (const float*)d_in[13];
  const float* b_msk3 = (const float*)d_in[14];
  const float* filt3  = (const float*)d_in[15];
  float* out = (float*)d_out;

  // ---- workspace layout ----
  float* om1 = (float*)d_ws;                       // [2, 27, HW] f32
  float* om2 = om1 + (size_t)NB * 3 * 9  * HW;     // [2, 75, HW]
  float* om3 = om2 + (size_t)NB * 3 * 25 * HW;     // [2,147, HW]
  float* xTf = om3 + (size_t)NB * 3 * 49 * HW;     // [2, HW, 64] f32
  __bf16* xTp = (__bf16*)(xTf + (size_t)NB * HW * CIN); // [2, HP, WP, 64] bf16
  __bf16* A1 = xTp + (size_t)NB * HP * WP * CIN;   // packed off/mask weights
  __bf16* A2 = A1 + (size_t)2  * (64 * 9)  * 16;
  __bf16* A3 = A2 + (size_t)5  * (64 * 25) * 16;
  __bf16* F1 = A3 + (size_t)10 * (64 * 49) * 16;   // packed filters
  __bf16* F2 = F1 + (size_t)2  * (64 * 9)  * 16;
  __bf16* F3 = F2 + (size_t)2  * (64 * 25) * 16;

  // ---- one-shot repacks ----
  pad_transpose_kernel<<<(NB * HP * WP * CIN + 255) / 256, 256, 0, stream>>>(x, xTf, xTp);
  wpack_kernel<3><<<(2 * (64 * 9) * 16 + 255) / 256, 256, 0, stream>>>(w_off1, w_msk1, A1);
  wpack_kernel<5><<<(5 * (64 * 25) * 16 + 255) / 256, 256, 0, stream>>>(w_off2, w_msk2, A2);
  wpack_kernel<7><<<(10 * (64 * 49) * 16 + 255) / 256, 256, 0, stream>>>(w_off3, w_msk3, A3);
  fpack_kernel<3><<<(2 * (64 * 9) * 16 + 255) / 256, 256, 0, stream>>>(filt1, F1);
  fpack_kernel<5><<<(2 * (64 * 25) * 16 + 255) / 256, 256, 0, stream>>>(filt2, F2);
  fpack_kernel<7><<<(2 * (64 * 49) * 16 + 255) / 256, 256, 0, stream>>>(filt3, F3);

  // ---- offset/mask convs (WMMA GEMMs); tile counts are multiples of 8 ----
  offmask_conv_kernel<3><<<2  * (NPIX / 32) / 8, 256, 0, stream>>>(
      xTp, (const v16bf*)A1, b_off1, b_msk1, om1);
  offmask_conv_kernel<5><<<5  * (NPIX / 32) / 8, 256, 0, stream>>>(
      xTp, (const v16bf*)A2, b_off2, b_msk2, om2);
  offmask_conv_kernel<7><<<10 * (NPIX / 32) / 8, 256, 0, stream>>>(
      xTp, (const v16bf*)A3, b_off3, b_msk3, om3);

  // ---- deformable convs (bilinear gather + WMMA GEMM) ----
  deform_gemm_kernel<3><<<NPIX / 32, 64, 0, stream>>>(xTf, om1, (const v16bf*)F1, out, 0);
  deform_gemm_kernel<5><<<NPIX / 32, 64, 0, stream>>>(xTf, om2, (const v16bf*)F2, out, 32);
  deform_gemm_kernel<7><<<NPIX / 32, 64, 0, stream>>>(xTf, om3, (const v16bf*)F3, out, 64);
}
